// GeneratorBlock_61418032333433
// MI455X (gfx1250) — matrix operand order repaired and checked by reading, not compiled, not run on hardware
//
#include <hip/hip_runtime.h>

#define HH   512
#define WW   512
#define CIN  64
#define COUT 64
#define LAT  512
#define KDIM 576   // CIN * 9 taps

typedef __attribute__((ext_vector_type(16))) __bf16 v16bf;
typedef __attribute__((ext_vector_type(8)))  __bf16 v8bf;
typedef __attribute__((ext_vector_type(8)))  float  v8f;

// CDNA5 async global->LDS path (ASYNCcnt) if the toolchain exposes it.
#if defined(__has_builtin)
#if __has_builtin(__builtin_amdgcn_global_load_async_to_lds_b128)
#define USE_ASYNC_LDS 1
#endif
#endif
#ifndef USE_ASYNC_LDS
#define USE_ASYNC_LDS 0
#endif

#if USE_ASYNC_LDS
// Probe-verified: param 1 = AS(1) int4* (global src), param 2 = LDS dst.
typedef int v4i_vs __attribute__((vector_size(16)));
typedef __attribute__((address_space(1))) v4i_vs* g_v4i_p;
typedef __attribute__((address_space(3))) v4i_vs* l_v4i_p;
#endif

__device__ __forceinline__ float leaky(float v) { return v > 0.f ? v : 0.2f * v; }

// ---------------------------------------------------------------------------
// Kernel 1: style vector, kw-indexed demod factors, modulated bf16 weights.
// wmc layout: [oc][tap][cin]  (64 x 9 x 64 bf16)
// wmr layout: [oc(16, rows 3..15 zero)][tap][cin]
// ---------------------------------------------------------------------------
__global__ __launch_bounds__(256) void prep_kernel(
    const float* __restrict__ sv, const float* __restrict__ w_style,
    const float* __restrict__ b_style, const float* __restrict__ w_conv,
    const float* __restrict__ w_rgb, __bf16* __restrict__ wmc,
    __bf16* __restrict__ wmr)
{
  __shared__ float style[CIN];
  __shared__ float psum[192];
  __shared__ float psumr[9];
  __shared__ float dconv[3];
  __shared__ float drgb[3];
  const int t = threadIdx.x;

  if (t < CIN) {
    float acc = b_style[t];
    for (int l = 0; l < LAT; ++l) acc += sv[l] * w_style[t * LAT + l];
    style[t] = acc;
  }
  __syncthreads();

  // Reference demod sums axes (O, I, kh) -> factor indexed by kw.
  if (t < 192) {
    const int kw = t / 64, o = t % 64;
    float ss = 0.f;
    for (int i = 0; i < CIN; ++i) {
      const float s = style[i];
      for (int kh = 0; kh < 3; ++kh) {
        const float w = w_conv[((o * CIN + i) * 3 + kh) * 3 + kw] * s;
        ss += w * w;
      }
    }
    psum[t] = ss;
  } else if (t < 201) {
    const int t2 = t - 192, kw = t2 / 3, o = t2 % 3;
    float ss = 0.f;
    for (int i = 0; i < CIN; ++i) {
      const float s = style[i];
      for (int kh = 0; kh < 3; ++kh) {
        const float w = w_rgb[((o * CIN + i) * 3 + kh) * 3 + kw] * s;
        ss += w * w;
      }
    }
    psumr[t2] = ss;
  }
  __syncthreads();

  if (t < 3) {
    float ss = 0.f;
    for (int o = 0; o < 64; ++o) ss += psum[t * 64 + o];
    dconv[t] = rsqrtf(ss + 1e-8f);
  } else if (t < 6) {
    const int kw = t - 3;
    drgb[kw] = rsqrtf(psumr[kw * 3] + psumr[kw * 3 + 1] + psumr[kw * 3 + 2] + 1e-8f);
  }
  __syncthreads();

  for (int e = t; e < COUT * KDIM; e += 256) {
    const int oc = e / KDIM, k = e % KDIM, tap = k >> 6, c = k & 63;
    wmc[e] = (__bf16)(w_conv[(oc * CIN + c) * 9 + tap] * style[c] * dconv[tap % 3]);
  }
  for (int e = t; e < 16 * KDIM; e += 256) {
    const int oc = e / KDIM, k = e % KDIM, tap = k >> 6, c = k & 63;
    float v = 0.f;
    if (oc < 3) v = w_rgb[(oc * CIN + c) * 9 + tap] * style[c] * drgb[tap % 3];
    wmr[e] = (__bf16)v;
  }
}

// Load one 32-wide B fragment pair (two rows, same pixel/lane) and issue two
// chained WMMAs on independent accumulators so LDS latency overlaps WMMA.
#define B_PAIR_WMMA(rowbase, afr, acc0, acc1)                                  \
  do {                                                                         \
    const v8bf b0a = ((const v8bf*)(rowbase))[0];                              \
    const v8bf b0b = ((const v8bf*)(rowbase))[1];                              \
    const v8bf b1a = *(const v8bf*)((rowbase) + 18 * CIN);                     \
    const v8bf b1b = *(const v8bf*)((rowbase) + 18 * CIN + 8);                 \
    v16bf bv0, bv1;                                                            \
    _Pragma("unroll") for (int i = 0; i < 8; ++i) {                            \
      bv0[i] = b0a[i]; bv0[i + 8] = b0b[i];                                    \
      bv1[i] = b1a[i]; bv1[i + 8] = b1b[i];                                    \
    }                                                                          \
    (acc0) = __builtin_amdgcn_wmma_f32_16x16x32_bf16(                          \
        false, (afr), false, bv0, (short)0, (acc0), false, false);             \
    (acc1) = __builtin_amdgcn_wmma_f32_16x16x32_bf16(                          \
        false, (afr), false, bv1, (short)0, (acc1), false, false);             \
  } while (0)

// ---------------------------------------------------------------------------
// Kernel 2: main modulated conv, implicit GEMM via v_wmma_f32_16x16x32_bf16.
// Block = 16x16 output tile, 8 waves: (oc-tile = wave&3, row-half = wave>>2).
// A = weights (M=16 oc x K32) held in VGPRs; B = activations (K32 x N=16 px)
// from LDS patch [y][x][c]. Rows processed in pairs for latency overlap.
// ---------------------------------------------------------------------------
__global__ __launch_bounds__(256) void conv_main_kernel(
    const float* __restrict__ image, const float* __restrict__ noise,
    const float* __restrict__ w_noise, const float* __restrict__ b_noise,
    const __bf16* __restrict__ wmc, float* __restrict__ img_out,
    __bf16* __restrict__ imgb)
{
  __shared__ __align__(16) __bf16 patch[18 * 18 * CIN];
  const int tid = threadIdx.x;
  const int w0 = blockIdx.x * 16, h0 = blockIdx.y * 16;

  const int wave = tid >> 5, lane = tid & 31;
  const int oc0 = (wave & 3) * 16, rowgrp = wave >> 2;
  const int m = lane & 15, hi = lane >> 4;

  const v8bf* wrow = (const v8bf*)(wmc + (oc0 + m) * KDIM);
  __builtin_prefetch(wrow, 0, 0);  // global_prefetch_b8: warm weight row

  // Stage input patch (with zero halo) as bf16, [y][x][c].
  for (int e = tid; e < 18 * 18 * CIN; e += 256) {
    const int x = e % 18, t2 = e / 18, y = t2 % 18, c = t2 / 18;
    const int gx = w0 - 1 + x, gy = h0 - 1 + y;
    float v = 0.f;
    if ((unsigned)gx < WW && (unsigned)gy < HH) v = image[(c * HH + gy) * WW + gx];
    patch[(y * 18 + x) * CIN + c] = (__bf16)v;
  }

  // A fragments: lane m holds weight row oc0+m. k order used here (contiguous, same as B):
  // lanes<16: K 0..15; lanes>=16: K 16..31.
  v16bf afrag[18];
  #pragma unroll
  for (int kc = 0; kc < 18; ++kc) {
    const int base = (kc >> 1) * 8 + (kc & 1) * 4;  // v8bf units
    const v8bf lo = wrow[base + 2 * hi];
    const v8bf hv = wrow[base + 2 * hi + 1];
    v16bf a;
    #pragma unroll
    for (int i = 0; i < 8; ++i) { a[i] = lo[i]; a[i + 8] = hv[i]; }
    afrag[kc] = a;
  }

  // Row-invariant epilogue constants (oc per lane/VGPR fixed across rows).
  float wn[8], bn[8];
  #pragma unroll
  for (int r = 0; r < 8; ++r) {
    const int oc = oc0 + r + hi * 8;
    wn[r] = w_noise[oc];
    bn[r] = b_noise[oc];
  }
  __syncthreads();

  #pragma unroll 1
  for (int rp = 0; rp < 4; ++rp) {
    const int y0 = rowgrp * 8 + rp * 2;
    v8f acc0 = {}, acc1 = {};
    #pragma unroll
    for (int kc = 0; kc < 18; ++kc) {
      const int tap = kc >> 1, kh = tap / 3, kw = tap % 3;
      const int c0 = (kc & 1) * 32 + hi * 16;  // B: lanes<16 K0..15, >=16 K16..31
      const __bf16* rowbase = &patch[((y0 + kh) * 18 + (m + kw)) * CIN + c0];
      B_PAIR_WMMA(rowbase, afrag[kc], acc0, acc1);
    }
    // D layout: lane%16 = N (pixel), VGPR r -> M = r + 8*(lane/16) (oc).
    const int gx = w0 + m;
    #pragma unroll
    for (int half = 0; half < 2; ++half) {
      const v8f& acc = half ? acc1 : acc0;
      const int gy = h0 + y0 + half;
      const float nz = noise[gy * WW + gx];
      v8bf pk;
      #pragma unroll
      for (int r = 0; r < 8; ++r) {
        float v = leaky(acc[r]);
        v = leaky(v + nz * wn[r] + bn[r]);
        const int oc = oc0 + r + hi * 8;
        img_out[(size_t)(oc * HH + gy) * WW + gx] = v;
        pk[r] = (__bf16)v;
      }
      if (imgb) {  // bf16 channel-last relay for the toRGB pass (one b128 store)
        *(v8bf*)&imgb[((size_t)gy * WW + gx) * CIN + oc0 + hi * 8] = pk;
      }
    }
  }
}

// ---------------------------------------------------------------------------
// Kernel 3 (fast): toRGB conv reading the bf16 channel-last relay. Patch
// staging is a pure 16B copy -> async global->LDS when available.
// ---------------------------------------------------------------------------
__global__ __launch_bounds__(256) void conv_rgb_fast_kernel(
    const __bf16* __restrict__ imgb, const float* __restrict__ prev_rgb,
    const __bf16* __restrict__ wmr, float* __restrict__ rgb_out)
{
  __shared__ __align__(16) __bf16 patch[18 * 18 * CIN];
  const int tid = threadIdx.x;
  const int w0 = blockIdx.x * 16, h0 = blockIdx.y * 16;
  const int wave = tid >> 5, lane = tid & 31;
  const int m = lane & 15, hi = lane >> 4;

  const v8bf* wrow = (const v8bf*)(wmr + m * KDIM);
  __builtin_prefetch(wrow, 0, 0);

  for (int e = tid; e < 18 * 18 * 8; e += 256) {
    const int c8 = (e & 7) * 8;
    const int p = e >> 3;
    const int x = p % 18, y = p / 18;
    const int gx = w0 - 1 + x, gy = h0 - 1 + y;
    __bf16* dst = &patch[(y * 18 + x) * CIN + c8];
    if ((unsigned)gx < WW && (unsigned)gy < HH) {
      const __bf16* src = &imgb[((size_t)gy * WW + gx) * CIN + c8];
#if USE_ASYNC_LDS
      __builtin_amdgcn_global_load_async_to_lds_b128(
          (g_v4i_p)const_cast<__bf16*>(src), (l_v4i_p)dst, 0, 0);
#else
      *(v8bf*)dst = *(const v8bf*)src;
#endif
    } else {
      v8bf z = {};
      *(v8bf*)dst = z;
    }
  }
#if USE_ASYNC_LDS
#if __has_builtin(__builtin_amdgcn_s_wait_asynccnt)
  __builtin_amdgcn_s_wait_asynccnt(0);
#else
  asm volatile("s_wait_asynccnt 0x0" ::: "memory");
#endif
#endif

  v16bf afrag[18];
  #pragma unroll
  for (int kc = 0; kc < 18; ++kc) {
    const int base = (kc >> 1) * 8 + (kc & 1) * 4;
    const v8bf lo = wrow[base + 2 * hi];
    const v8bf hv = wrow[base + 2 * hi + 1];
    v16bf a;
    #pragma unroll
    for (int i = 0; i < 8; ++i) { a[i] = lo[i]; a[i + 8] = hv[i]; }
    afrag[kc] = a;
  }
  __syncthreads();

  const int y0 = wave * 2;
  v8f acc0 = {}, acc1 = {};
  #pragma unroll
  for (int kc = 0; kc < 18; ++kc) {
    const int tap = kc >> 1, kh = tap / 3, kw = tap % 3;
    const int c0 = (kc & 1) * 32 + hi * 16;
    const __bf16* rowbase = &patch[((y0 + kh) * 18 + (m + kw)) * CIN + c0];
    B_PAIR_WMMA(rowbase, afrag[kc], acc0, acc1);
  }
  if (hi == 0) {  // oc rows 0..2 live in lanes 0..15, VGPRs 0..2
    const int gx = w0 + m;
    #pragma unroll
    for (int half = 0; half < 2; ++half) {
      const v8f& acc = half ? acc1 : acc0;
      const int gy = h0 + y0 + half;
      #pragma unroll
      for (int r = 0; r < 3; ++r) {
        const size_t o = (size_t)(r * HH + gy) * WW + gx;
        rgb_out[o] = leaky(acc[r]) + prev_rgb[o];
      }
    }
  }
}

// ---------------------------------------------------------------------------
// Kernel 3 (fallback): toRGB conv reading img as f32 (small-ws path).
// ---------------------------------------------------------------------------
__global__ __launch_bounds__(256) void conv_rgb_kernel(
    const float* __restrict__ img, const float* __restrict__ prev_rgb,
    const __bf16* __restrict__ wmr, float* __restrict__ rgb_out)
{
  __shared__ __align__(16) __bf16 patch[18 * 18 * CIN];
  const int tid = threadIdx.x;
  const int w0 = blockIdx.x * 16, h0 = blockIdx.y * 16;

  for (int e = tid; e < 18 * 18 * CIN; e += 256) {
    const int x = e % 18, t2 = e / 18, y = t2 % 18, c = t2 / 18;
    const int gx = w0 - 1 + x, gy = h0 - 1 + y;
    float v = 0.f;
    if ((unsigned)gx < WW && (unsigned)gy < HH) v = img[(c * HH + gy) * WW + gx];
    patch[(y * 18 + x) * CIN + c] = (__bf16)v;
  }
  const int wave = tid >> 5, lane = tid & 31;
  const int m = lane & 15, hi = lane >> 4;

  v16bf afrag[18];
  {
    const v8bf* wrow = (const v8bf*)(wmr + m * KDIM);
    #pragma unroll
    for (int kc = 0; kc < 18; ++kc) {
      const int base = (kc >> 1) * 8 + (kc & 1) * 4;
      const v8bf lo = wrow[base + 2 * hi];
      const v8bf hv = wrow[base + 2 * hi + 1];
      v16bf a;
      #pragma unroll
      for (int i = 0; i < 8; ++i) { a[i] = lo[i]; a[i + 8] = hv[i]; }
      afrag[kc] = a;
    }
  }
  __syncthreads();

  const int y0 = wave * 2;
  v8f acc0 = {}, acc1 = {};
  #pragma unroll
  for (int kc = 0; kc < 18; ++kc) {
    const int tap = kc >> 1, kh = tap / 3, kw = tap % 3;
    const int c0 = (kc & 1) * 32 + hi * 16;
    const __bf16* rowbase = &patch[((y0 + kh) * 18 + (m + kw)) * CIN + c0];
    B_PAIR_WMMA(rowbase, afrag[kc], acc0, acc1);
  }
  if (hi == 0) {
    const int gx = w0 + m;
    #pragma unroll
    for (int half = 0; half < 2; ++half) {
      const v8f& acc = half ? acc1 : acc0;
      const int gy = h0 + y0 + half;
      #pragma unroll
      for (int r = 0; r < 3; ++r) {
        const size_t o = (size_t)(r * HH + gy) * WW + gx;
        rgb_out[o] = leaky(acc[r]) + prev_rgb[o];
      }
    }
  }
}

// ---------------------------------------------------------------------------
extern "C" void kernel_launch(void* const* d_in, const int* in_sizes, int n_in,
                              void* d_out, int out_size, void* d_ws, size_t ws_size,
                              hipStream_t stream) {
  const float* image    = (const float*)d_in[0];
  const float* prev_rgb = (const float*)d_in[1];
  const float* sv       = (const float*)d_in[2];
  const float* noise    = (const float*)d_in[3];
  const float* w_conv   = (const float*)d_in[4];
  const float* w_rgb    = (const float*)d_in[5];
  const float* w_noise  = (const float*)d_in[6];
  const float* b_noise  = (const float*)d_in[7];
  const float* w_style  = (const float*)d_in[8];
  const float* b_style  = (const float*)d_in[9];

  float* img_out = (float*)d_out;
  float* rgb_out = img_out + (size_t)COUT * HH * WW;

  __bf16* wmc = (__bf16*)d_ws;              // 64*576 bf16
  __bf16* wmr = wmc + (size_t)COUT * KDIM;  // 16*576 bf16 (rows 3..15 zero)

  const size_t wbytes = (size_t)(COUT + 16) * KDIM * sizeof(__bf16);  // 92160
  const size_t imgb_bytes = (size_t)HH * WW * CIN * sizeof(__bf16);   // 32 MiB
  const bool fast = (ws_size >= wbytes + imgb_bytes);
  __bf16* imgb = fast ? (__bf16*)((char*)d_ws + wbytes) : nullptr;

  prep_kernel<<<1, 256, 0, stream>>>(sv, w_style, b_style, w_conv, w_rgb, wmc, wmr);

  dim3 grid(WW / 16, HH / 16);
  conv_main_kernel<<<grid, 256, 0, stream>>>(image, noise, w_noise, b_noise, wmc,
                                             img_out, imgb);
  if (fast)
    conv_rgb_fast_kernel<<<grid, 256, 0, stream>>>(imgb, prev_rgb, wmr, rgb_out);
  else
    conv_rgb_kernel<<<grid, 256, 0, stream>>>(img_out, prev_rgb, wmr, rgb_out);
}
